// MultiLayerVQ_18468359373177
// MI455X (gfx1250) — compile-verified, matrix operand order checked by reading, not run on hardware
//
#include <hip/hip_runtime.h>

// MI455X / gfx1250 multi-layer VQ-VAE quantization.
// Layers N=4, codes K=1024, dim D=128, rows M=B*H*W=32768.
// Outputs (concat, float): quantized[16777216], indices[131072], loss[4], perp[4].

typedef __attribute__((ext_vector_type(16))) __bf16 v16bf;
typedef __attribute__((ext_vector_type(8)))  float  v8f;

#define NLAYER 4
#define KCODES 1024
#define DDIM   128
#define MROWS  32768          // 32 * 32 * 32
#define HW     1024           // 32*32
#define XLAYER (DDIM * HW)    // 131072 floats per (b, layer)
#define XBATCH (NLAYER * XLAYER) // 524288 floats per batch image

// ws layout (dwords):
//   [0 .. 524287]        bf16 hi/lo codebook B tiles: (n,kc,dc) tile = 512 dwords
//                        (dwords 0-255: hi, [lane][v]=lane*8+v ; 256-511: lo)
//   [524288 .. 528383]   cbnorm: ||e||^2 per (n,k), fp32
//   [528384 .. 532479]   counts: per (n,k), fp32
#define WS_B_DWORDS  (NLAYER * 64 * 4 * 512)

typedef __attribute__((address_space(3))) unsigned lds_uint;

// Async global->LDS copy, 16 bytes per lane, tracked by ASYNCcnt.
// VFLAT GLOBAL_LOAD_ASYNC_TO_LDS_B128 (opcode 98), GV addressing mode.
__device__ __forceinline__ void async_copy_b128(const unsigned* gsrc, unsigned* ldst) {
    unsigned lofs = (unsigned)(unsigned long long)(lds_uint*)ldst;  // 32-bit LDS addr
    asm volatile("global_load_async_to_lds_b128 %0, %1, off"
                 :: "v"(lofs), "v"(gsrc)
                 : "memory");
}

__device__ __forceinline__ void wait_asynccnt0() {
#if defined(__has_builtin) && __has_builtin(__builtin_amdgcn_s_wait_asynccnt)
    __builtin_amdgcn_s_wait_asynccnt(0);
#else
    asm volatile("s_wait_asynccnt 0" ::: "memory");
#endif
}

__device__ __forceinline__ void split_bf16(float x, unsigned& hi, unsigned& lo) {
    unsigned u = __float_as_uint(x);
    unsigned h = (u + 0x7FFFu + ((u >> 16) & 1u)) >> 16;   // RNE to bf16
    float hf = __uint_as_float(h << 16);
    float r = x - hf;
    unsigned ur = __float_as_uint(r);
    unsigned l = (ur + 0x7FFFu + ((ur >> 16) & 1u)) >> 16;
    hi = h; lo = l;
}

// ---------------------------------------------------------------------------
// Kernel 0: zero counts and loss accumulators (d_out is poisoned, not zeroed).
__global__ void vq_init(float* __restrict__ counts, float* __restrict__ out_loss) {
    int t = blockIdx.x * 256 + threadIdx.x;
    if (t < NLAYER * KCODES) counts[t] = 0.0f;
    if (t < NLAYER)          out_loss[t] = 0.0f;
}

// ---------------------------------------------------------------------------
// Kernel 1: codebook -> bf16 hi/lo in WMMA B-tile layout + fp32 ||e||^2.
// B tile for (n,kc,dc) is 32(d) x 16(code) bf16:
//   lane = h*16 + (k%16), vgpr v holds packed d-pair:
//   d_local = (v>=4 ? 16 : 0) + (v%4)*2 + h*8, d = dc*32 + d_local.
__global__ void vq_prep(const float* __restrict__ cb,
                        unsigned* __restrict__ wsb,
                        float* __restrict__ cbnorm) {
    int t = blockIdx.x * 256 + threadIdx.x;
    if (t >= NLAYER * KCODES) return;
    int n = t >> 10, k = t & 1023;
    int kc = k >> 4, kk = k & 15;
    const float* row = cb + (size_t)t * DDIM;
    float norm = 0.0f;
    for (int d = 0; d < DDIM; d += 2) {
        float e0 = row[d], e1 = row[d + 1];
        norm += e0 * e0 + e1 * e1;
        unsigned h0, l0, h1, l1;
        split_bf16(e0, h0, l0);
        split_bf16(e1, h1, l1);
        unsigned hp = h0 | (h1 << 16);
        unsigned lp = l0 | (l1 << 16);
        int dc = d >> 5, dd = d & 31;
        int v4 = (dd >> 4) & 1, dd2 = dd & 15;
        int hh = dd2 >> 3, rem = dd2 & 7, vm = rem >> 1;
        int v = v4 * 4 + vm;
        int lane = hh * 16 + kk;
        size_t base = ((size_t)((n * 64 + kc) * 4 + dc)) * 512 + (size_t)lane * 8 + v;
        wsb[base]       = hp;
        wsb[base + 256] = lp;
    }
    cbnorm[t] = norm;
}

// ---------------------------------------------------------------------------
// Kernel 2: per-wave 16-row tile; B tiles staged in LDS (shared by 8 waves,
// double-buffered, async global->LDS copies overlapped with WMMA compute).
__global__ __launch_bounds__(256)
void vq_main(const float* __restrict__ x, const float* __restrict__ cb,
             const unsigned* __restrict__ wsb, const float* __restrict__ cbnorm,
             float* __restrict__ counts,
             float* __restrict__ out_q, float* __restrict__ out_idx,
             float* __restrict__ out_loss) {
    __shared__ unsigned sB[2][2048];   // double-buffered kc tile: 4 dc x 512 dw
    __shared__ float    sNorm[KCODES];
    __shared__ int      s_idx[8][16];

    const int n    = blockIdx.y;
    const int lane = threadIdx.x & 31;
    const int wave = threadIdx.x >> 5;
    const int hh   = lane >> 4;       // half-wave
    const int lk   = lane & 15;
    const int m_base = (blockIdx.x * 8 + wave) * 16;
    const int mrow = m_base + lk;
    const int bb = mrow >> 10, hw = mrow & 1023;
    const size_t xrow = (size_t)bb * XBATCH + (size_t)n * XLAYER + hw;

    // --- Stage ||e||^2 row for this layer in LDS.
    for (int i = threadIdx.x; i < KCODES; i += 256)
        sNorm[i] = cbnorm[n * KCODES + i];

    // --- Load A tile (16 rows x 128 d) once: bf16 hi/lo packed, WMMA A layout.
    unsigned a_hi[4][8], a_lo[4][8];
#pragma unroll
    for (int dc = 0; dc < 4; ++dc) {
#pragma unroll
        for (int v = 0; v < 8; ++v) {
            int dl = ((v >= 4) ? 16 : 0) + (v & 3) * 2 + hh * 8;
            int d = dc * 32 + dl;
            float f0 = x[xrow + (size_t)d * HW];
            float f1 = x[xrow + (size_t)(d + 1) * HW];
            unsigned h0, l0, h1, l1;
            split_bf16(f0, h0, l0);
            split_bf16(f1, h1, l1);
            a_hi[dc][v] = h0 | (h1 << 16);
            a_lo[dc][v] = l0 | (l1 << 16);
        }
    }

    float best[8];
    int   bidx[8];
#pragma unroll
    for (int r = 0; r < 8; ++r) { best[r] = 3.4e38f; bidx[r] = 0; }

    const unsigned* wb = wsb + (size_t)n * 64 * 2048;  // this layer's tile stream
    const int c0 = threadIdx.x * 2;                    // 2 x b128 chunks / thread

    // Preload kc=0 into buffer 0 via async global->LDS.
#pragma unroll
    for (int c = 0; c < 2; ++c)
        async_copy_b128(wb + (size_t)(c0 + c) * 4, &sB[0][(c0 + c) * 4]);
    wait_asynccnt0();
    __syncthreads();

    // --- Scan K=1024 codes, 16 at a time. score = ||e||^2 - 2 x.e
    for (int kc = 0; kc < 64; ++kc) {
        const int cur = kc & 1;
        if (kc < 63) {
            const unsigned* src = wb + (size_t)(kc + 1) * 2048;
#pragma unroll
            for (int c = 0; c < 2; ++c)
                async_copy_b128(src + (size_t)(c0 + c) * 4,
                                &sB[cur ^ 1][(c0 + c) * 4]);
        }
        if (kc < 62)
            __builtin_prefetch(wb + (size_t)(kc + 2) * 2048 + threadIdx.x * 8, 0, 3);

        float en2 = sNorm[kc * 16 + lk];
        v8f acc = {};
        const unsigned* bt = &sB[cur][0];
#pragma unroll
        for (int dc = 0; dc < 4; ++dc) {
            uint4 bh0 = *(const uint4*)(bt + dc * 512 + lane * 8);
            uint4 bh1 = *(const uint4*)(bt + dc * 512 + lane * 8 + 4);
            uint4 bl0 = *(const uint4*)(bt + dc * 512 + 256 + lane * 8);
            uint4 bl1 = *(const uint4*)(bt + dc * 512 + 256 + lane * 8 + 4);

            union { unsigned u[8]; v16bf v; } BH, BL, AH, AL;
            BH.u[0] = bh0.x; BH.u[1] = bh0.y; BH.u[2] = bh0.z; BH.u[3] = bh0.w;
            BH.u[4] = bh1.x; BH.u[5] = bh1.y; BH.u[6] = bh1.z; BH.u[7] = bh1.w;
            BL.u[0] = bl0.x; BL.u[1] = bl0.y; BL.u[2] = bl0.z; BL.u[3] = bl0.w;
            BL.u[4] = bl1.x; BL.u[5] = bl1.y; BL.u[6] = bl1.z; BL.u[7] = bl1.w;
#pragma unroll
            for (int i = 0; i < 8; ++i) { AH.u[i] = a_hi[dc][i]; AL.u[i] = a_lo[dc][i]; }

            // fp32-ish dot via 3-term bf16 split: lo*hi + hi*lo + hi*hi
            acc = __builtin_amdgcn_wmma_f32_16x16x32_bf16(false, AL.v, false, BH.v,
                                                          (short)0, acc, false, false);
            acc = __builtin_amdgcn_wmma_f32_16x16x32_bf16(false, AH.v, false, BL.v,
                                                          (short)0, acc, false, false);
            acc = __builtin_amdgcn_wmma_f32_16x16x32_bf16(false, AH.v, false, BH.v,
                                                          (short)0, acc, false, false);
        }
        int kcur = kc * 16 + lk;
#pragma unroll
        for (int r = 0; r < 8; ++r) {
            float s = en2 - 2.0f * acc[r];
            if (s < best[r]) { best[r] = s; bidx[r] = kcur; }
        }

        if (kc < 63) {
            wait_asynccnt0();   // this wave's share of the next tile is in LDS
            __syncthreads();    // ... and everyone else's share, too
        }
    }

    // --- Argmin across the 16 code-columns (lanes within each half-wave).
#pragma unroll
    for (int r = 0; r < 8; ++r) {
#pragma unroll
        for (int off = 1; off < 16; off <<= 1) {
            float ov = __shfl_xor(best[r], off, 32);
            int   oi = __shfl_xor(bidx[r], off, 32);
            if (ov < best[r] || (ov == best[r] && oi < bidx[r])) {
                best[r] = ov; bidx[r] = oi;
            }
        }
    }

    // --- Emit indices, histogram counts, broadcast idx via LDS.
    if (lk == 0) {
#pragma unroll
        for (int r = 0; r < 8; ++r) {
            int m = m_base + r + 8 * hh;
            int mb = m >> 10, mhw = m & 1023;
            int ii = bidx[r];
            out_idx[(size_t)mb * (NLAYER * HW) + (size_t)n * HW + mhw] = (float)ii;
            atomicAdd(&counts[n * KCODES + ii], 1.0f);
            s_idx[wave][r + 8 * hh] = ii;
        }
    }
    __syncthreads();

    // --- Gather fp32 codebook rows, write quantized (straight-through fwd = q),
    //     and accumulate 1.25 * mean((q-x)^2).
    int myidx = s_idx[wave][lk];
    const float* qrow = cb + ((size_t)n * KCODES + myidx) * DDIM;
    float lsum = 0.0f;
    for (int j = 0; j < 64; ++j) {
        int d = j * 2 + hh;
        float qv = qrow[d];
        float xv = x[xrow + (size_t)d * HW];
        float df = qv - xv;
        lsum += df * df;
        out_q[xrow + (size_t)d * HW] = qv;
    }
#pragma unroll
    for (int off = 16; off >= 1; off >>= 1) lsum += __shfl_xor(lsum, off, 32);
    if (lane == 0)
        atomicAdd(&out_loss[n], lsum * (1.25f / ((float)MROWS * (float)DDIM)));
}

// ---------------------------------------------------------------------------
// Kernel 3: perplexity per layer from counts.
__global__ void vq_finalize(const float* __restrict__ counts,
                            float* __restrict__ out_perp) {
    __shared__ float red[256];
    int n = blockIdx.x, t = threadIdx.x;
    float s = 0.0f;
    for (int k = t; k < KCODES; k += 256) {
        float p = counts[n * KCODES + k] * (1.0f / (float)MROWS);
        s += p * __logf(p + 1e-10f);
    }
    red[t] = s;
    __syncthreads();
    for (int o = 128; o > 0; o >>= 1) {
        if (t < o) red[t] += red[t + o];
        __syncthreads();
    }
    if (t == 0) out_perp[n] = __expf(-red[0]);
}

// ---------------------------------------------------------------------------
extern "C" void kernel_launch(void* const* d_in, const int* in_sizes, int n_in,
                              void* d_out, int out_size, void* d_ws, size_t ws_size,
                              hipStream_t stream) {
    const float* x  = (const float*)d_in[0];   // [32, 512, 32, 32]
    const float* cb = (const float*)d_in[1];   // [4, 1024, 128]

    float* out      = (float*)d_out;
    float* out_q    = out;                               // 16,777,216
    float* out_idx  = out_q + (size_t)32 * XBATCH;       //    131,072
    float* out_loss = out_idx + (size_t)32 * NLAYER * HW;//          4
    float* out_perp = out_loss + NLAYER;                 //          4

    unsigned* wsb   = (unsigned*)d_ws;
    float* cbnorm   = (float*)d_ws + WS_B_DWORDS;
    float* counts   = cbnorm + NLAYER * KCODES;

    vq_init<<<16, 256, 0, stream>>>(counts, out_loss);
    vq_prep<<<16, 256, 0, stream>>>(cb, wsb, cbnorm);
    vq_main<<<dim3(256, NLAYER), 256, 0, stream>>>(x, cb, wsb, cbnorm, counts,
                                                   out_q, out_idx, out_loss);
    vq_finalize<<<NLAYER, 256, 0, stream>>>(counts, out_perp);
}
